// RNN_91250875171637
// MI455X (gfx1250) — matrix operand order repaired and checked by reading, not compiled, not run on hardware
//
#include <hip/hip_runtime.h>

// ---------------------------------------------------------------------------
// RNN scan on MI455X (gfx1250, wave32, WMMA)
//   h0 = p0 @ W_enc
//   h_t = relu(v_t @ W_in + h_{t-1} @ W_rec)      (t = 0..99, serial)
//   out[:,t,:] = h_t @ W_dec
// GEMMs via v_wmma_f32_16x16x32_f16 (f32 accumulate). Weights transposed +
// downconverted to f16 once into d_ws. The recurrent GEMM register-blocks
// 4 N-tiles per wave and stages the shared B strip through LDS with
// double-buffered GLOBAL_LOAD_ASYNC_TO_LDS_B128 (s_wait_asynccnt drained).
// Decoder of h_{t-1} is fused into the step-t launch (both read h_{t-1}).
// ---------------------------------------------------------------------------

typedef __attribute__((ext_vector_type(16))) _Float16 v16h;
typedef __attribute__((ext_vector_type(8)))  _Float16 v8h;
typedef __attribute__((ext_vector_type(8)))  float    v8f;
typedef __attribute__((ext_vector_type(4)))  int      v4i;

#define AS1 __attribute__((address_space(1)))
#define AS3 __attribute__((address_space(3)))

#define B_  256
#define T_  100
#define NG_ 2048
#define NP_ 512

#define LDS_ROW 40  // halfs per 32-half B row (80B: 16B-aligned, bank-spread)

#if __has_builtin(__builtin_amdgcn_global_load_async_to_lds_b128)
#define HAS_ASYNC 1
#else
#define HAS_ASYNC 0
#endif

#if HAS_ASYNC
#if __has_builtin(__builtin_amdgcn_s_wait_asynccnt)
#define WAIT_ASYNC(n) __builtin_amdgcn_s_wait_asynccnt(n)
#else
#define WAIT_ASYNC(n) asm volatile("s_wait_asynccnt %0" ::"n"(n) : "memory")
#endif
#else
#define WAIT_ASYNC(n)
#endif

// one 16-byte chunk: global -> LDS (async if available)
__device__ __forceinline__ void copy16(const _Float16* __restrict__ g,
                                       _Float16* __restrict__ l) {
#if HAS_ASYNC
    __builtin_amdgcn_global_load_async_to_lds_b128(
        (AS1 v4i*)g, (AS3 v4i*)l, 0, 0);
#else
    *(v8h*)l = *(const v8h*)g;
#endif
}

__device__ __forceinline__ v16h pack16(v8h lo, v8h hi) {
    v16h r;
#pragma unroll
    for (int i = 0; i < 8; ++i) { r[i] = lo[i]; r[8 + i] = hi[i]; }
    return r;
}

__device__ __forceinline__ v8f wmma32(v16h a, v16h b, v8f c) {
    return __builtin_amdgcn_wmma_f32_16x16x32_f16(false, a, false, b,
                                                  (short)0, c, false, false);
}

// ---- setup: f32 [R,C] -> f16 [C,R] (transpose + downconvert) ----
__global__ void cvt_transpose_f32_f16(const float* __restrict__ in,
                                      _Float16* __restrict__ out,
                                      int R, int C) {
    int idx = blockIdx.x * blockDim.x + threadIdx.x;
    if (idx >= R * C) return;
    int r = idx / C, c = idx % C;
    out[(size_t)c * R + r] = (_Float16)in[idx];
}

// ---- setup: flat f32 -> f16 ----
__global__ void cvt_f32_f16(const float* __restrict__ in,
                            _Float16* __restrict__ out, int n) {
    int idx = blockIdx.x * blockDim.x + threadIdx.x;
    if (idx < n) out[idx] = (_Float16)in[idx];
}

// ---------------------------------------------------------------------------
// Single 16x16 tile (used by encoder). Fragment layouts used here (contiguous K, same for A and B):
//   A (16x32): lane l<16 holds row l, K = [16*hi..+16)
//   B (32x16): lane l<16 holds col l, K = [16*hi..+16)
//   C/D      : VGPR j <-> (M = j + 8*hi, N = l)
// ---------------------------------------------------------------------------
__device__ __forceinline__ v8f tile_gemm(const _Float16* __restrict__ A, int lda,
                                         const _Float16* __restrict__ Bt, int ldb,
                                         int K, int lane) {
    const int l = lane & 15, hi = lane >> 4;
    const _Float16* arow = A + (size_t)l * lda + 16 * hi;
    const _Float16* brow = Bt + (size_t)l * ldb + 16 * hi;
    v8f acc = {};
#pragma unroll 2
    for (int k = 0; k < K; k += 32) {
        v16h av = pack16(*(const v8h*)(arow + k), *(const v8h*)(arow + k + 8));
        v16h bv = pack16(*(const v8h*)(brow + k), *(const v8h*)(brow + k + 8));
        acc = wmma32(av, bv, acc);
    }
    return acc;
}

// ---- encoder: h0[256,2048] = f16(p0 @ W_enc), no relu ----
__global__ __launch_bounds__(256) void enc_kernel(
    const _Float16* __restrict__ p0h,    // [256,512]
    const _Float16* __restrict__ WencT,  // [2048,512]
    _Float16* __restrict__ h0) {         // [256,2048]
    const int wave = threadIdx.x >> 5, lane = threadIdx.x & 31;
    const int tile = blockIdx.x * 8 + wave;  // 2048 tiles
    int mt = tile >> 7, nt = tile & 127;
    v8f acc = tile_gemm(p0h + (size_t)mt * 16 * NP_, NP_,
                        WencT + (size_t)nt * 16 * NP_, NP_, NP_, lane);
    const int l = lane & 15, hi = lane >> 4;
    _Float16* o = h0 + (size_t)(mt * 16 + 8 * hi) * NG_ + nt * 16 + l;
#pragma unroll
    for (int j = 0; j < 8; ++j) o[(size_t)j * NG_] = (_Float16)acc[j];
}

// ---- decoder wave-job: 1 M-tile x 4 N-tiles, A-fragment reused 4x ----
__device__ __forceinline__ void dec_tile4(const _Float16* __restrict__ hin,
                                          const _Float16* __restrict__ WdecT,
                                          float* __restrict__ out,
                                          int tout, int job, int lane) {
    const int l = lane & 15, hi = lane >> 4;
    const int mt = job >> 3;          // 16 M-tiles
    const int nt0 = (job & 7) * 4;    // 8 groups of 4 N-tiles (N=512)
    const _Float16* Arow = hin + (size_t)(mt * 16 + l) * NG_ + 16 * hi;
    v8f acc[4] = {};
#pragma unroll 2
    for (int k = 0; k < NG_; k += 32) {
        v16h av = pack16(*(const v8h*)(Arow + k), *(const v8h*)(Arow + k + 8));
#pragma unroll
        for (int nt = 0; nt < 4; ++nt) {
            const _Float16* bp =
                WdecT + (size_t)((nt0 + nt) * 16 + l) * NG_ + k + 16 * hi;
            acc[nt] = wmma32(av, pack16(*(const v8h*)bp, *(const v8h*)(bp + 8)),
                             acc[nt]);
        }
    }
#pragma unroll
    for (int j = 0; j < 8; ++j) {
        int b = mt * 16 + 8 * hi + j;
#pragma unroll
        for (int nt = 0; nt < 4; ++nt)
            out[((size_t)b * T_ + tout) * NP_ + (nt0 + nt) * 16 + l] = acc[nt][j];
    }
}

// ---------------------------------------------------------------------------
// Fused: blocks 0..63  -> h_t = relu(vproj_t + h_{t-1} @ W_rec)
//        blocks 64..79 -> out[:,t-1,:] = h_{t-1} @ W_dec   (skipped at t==0)
// Recurrence macro-tile: 128M x 64N per block; wave w owns M-tile w and
// 4 N-tiles. B strip (64N x 32K) staged via double-buffered async copy to LDS.
// ---------------------------------------------------------------------------
__global__ __launch_bounds__(256) void step_dec_kernel(
    const _Float16* __restrict__ hin,    // h_{t-1} [256,2048] f16
    _Float16* __restrict__ hout,         // h_t     [256,2048] f16
    const _Float16* __restrict__ WrecT,  // [2048,2048] f16 (transposed)
    const _Float16* __restrict__ WdecT,  // [512,2048]  f16 (transposed)
    const float* __restrict__ v,         // [256,100,2] f32
    const float* __restrict__ Win,       // [2,2048]    f32
    float* __restrict__ out,             // [256,100,512] f32
    int t) {
    __shared__ _Float16 lbuf[2][64 * LDS_ROW];  // 10 KB
    const int wave = threadIdx.x >> 5, lane = threadIdx.x & 31;
    const int l = lane & 15, hi = lane >> 4;

    if (blockIdx.x >= 64) {  // ---- decoder of h_{t-1} ----
        if (t == 0) return;
        dec_tile4(hin, WdecT, out, t - 1, (blockIdx.x - 64) * 8 + wave, lane);
        return;
    }

    // ---- recurrence ----
    const int m0 = (blockIdx.x >> 5) * 128 + wave * 16;  // 2 M-strips
    const int n0 = (blockIdx.x & 31) * 64;               // 32 N-strips
    const _Float16* Arow = hin + (size_t)(m0 + l) * NG_ + 16 * hi;

    // cooperative copy mapping: 256 threads x 16B = 64 rows x 32 halfs
    const int tr = threadIdx.x >> 2;        // row 0..63
    const int tc = (threadIdx.x & 3) * 8;   // half offset 0/8/16/24
    const _Float16* gsrc = WrecT + (size_t)(n0 + tr) * NG_ + tc;
    _Float16* ldst[2] = {&lbuf[0][tr * LDS_ROW + tc], &lbuf[1][tr * LDS_ROW + tc]};

    v8f acc[4] = {};
    copy16(gsrc, ldst[0]);                  // prologue: k = 0 -> buf0
    for (int k = 0, i = 0; k < NG_; k += 32, ++i) {
        const _Float16* cur = &lbuf[i & 1][0];
        if (k + 32 < NG_) {
            copy16(gsrc + k + 32, ldst[(i + 1) & 1]);  // prefetch next chunk
            WAIT_ASYNC(1);                             // cur chunk landed
        } else {
            WAIT_ASYNC(0);
        }
        __syncthreads();                               // cur visible to block

        v16h av = pack16(*(const v8h*)(Arow + k), *(const v8h*)(Arow + k + 8));
#pragma unroll
        for (int nt = 0; nt < 4; ++nt) {
            const _Float16* bp = cur + (nt * 16 + l) * LDS_ROW + 16 * hi;
            acc[nt] = wmma32(av, pack16(*(const v8h*)bp, *(const v8h*)(bp + 8)),
                             acc[nt]);
        }
        __syncthreads();  // all waves done with cur before it is overwritten
    }

    // epilogue: + v_t @ W_in, relu, store f16
    float w0[4], w1[4];
#pragma unroll
    for (int nt = 0; nt < 4; ++nt) {
        int n = n0 + nt * 16 + l;
        w0[nt] = Win[n];
        w1[nt] = Win[NG_ + n];
    }
#pragma unroll
    for (int j = 0; j < 8; ++j) {
        int b = m0 + 8 * hi + j;
        float v0 = v[((size_t)b * T_ + t) * 2 + 0];
        float v1 = v[((size_t)b * T_ + t) * 2 + 1];
#pragma unroll
        for (int nt = 0; nt < 4; ++nt) {
            float x = acc[nt][j] + v0 * w0[nt] + v1 * w1[nt];
            hout[(size_t)b * NG_ + n0 + nt * 16 + l] = (_Float16)(x > 0.0f ? x : 0.0f);
        }
    }
}

// ---- final decoder for h_{T-1} ----
__global__ __launch_bounds__(256) void dec_kernel(
    const _Float16* __restrict__ hin,
    const _Float16* __restrict__ WdecT,
    float* __restrict__ out, int tout) {
    const int wave = threadIdx.x >> 5, lane = threadIdx.x & 31;
    dec_tile4(hin, WdecT, out, tout, blockIdx.x * 8 + wave, lane);
}

extern "C" void kernel_launch(void* const* d_in, const int* in_sizes, int n_in,
                              void* d_out, int out_size, void* d_ws, size_t ws_size,
                              hipStream_t stream) {
    const float* v    = (const float*)d_in[0];   // [256,100,2]
    const float* p0   = (const float*)d_in[1];   // [256,512]
    const float* Wenc = (const float*)d_in[2];   // [512,2048]
    const float* Win  = (const float*)d_in[3];   // [2,2048]
    const float* Wrec = (const float*)d_in[4];   // [2048,2048]
    const float* Wdec = (const float*)d_in[5];   // [2048,512]
    float* out = (float*)d_out;

    // workspace layout (f16), ~16 MB total
    _Float16* ws    = (_Float16*)d_ws;
    _Float16* WrecT = ws;                    size_t o = (size_t)NG_ * NG_;
    _Float16* WencT = ws + o;                o += (size_t)NG_ * NP_;
    _Float16* WdecT = ws + o;                o += (size_t)NP_ * NG_;
    _Float16* p0h   = ws + o;                o += (size_t)B_ * NP_;
    _Float16* h0    = ws + o;                o += (size_t)B_ * NG_;
    _Float16* hA    = ws + o;                o += (size_t)B_ * NG_;
    _Float16* hB    = ws + o;
    (void)ws_size; (void)in_sizes; (void)n_in; (void)out_size;

    // weight transpose + f16 downconvert (L2-resident, cheap)
    cvt_transpose_f32_f16<<<(NG_ * NG_ + 255) / 256, 256, 0, stream>>>(Wrec, WrecT, NG_, NG_);
    cvt_transpose_f32_f16<<<(NP_ * NG_ + 255) / 256, 256, 0, stream>>>(Wenc, WencT, NP_, NG_);
    cvt_transpose_f32_f16<<<(NG_ * NP_ + 255) / 256, 256, 0, stream>>>(Wdec, WdecT, NG_, NP_);
    cvt_f32_f16<<<(B_ * NP_ + 255) / 256, 256, 0, stream>>>(p0, p0h, B_ * NP_);

    // encoder
    enc_kernel<<<256, 256, 0, stream>>>(p0h, WencT, h0);

    // serial scan; decoder of h_{t-1} rides along in extra blocks
    const _Float16* hin = h0;
    for (int t = 0; t < T_; ++t) {
        _Float16* hout = (t & 1) ? hB : hA;
        step_dec_kernel<<<80, 256, 0, stream>>>(hin, hout, WrecT, WdecT, v, Win, out, t);
        hin = hout;
    }
    dec_kernel<<<16, 256, 0, stream>>>(hin, WdecT, out, T_ - 1);
}
